// Encoder_17918603559377
// MI455X (gfx1250) — matrix operand order repaired and checked by reading, not compiled, not run on hardware
//
#include <hip/hip_runtime.h>
#include <hip/hip_bf16.h>

// ---------------------------------------------------------------------------
// Transformer encoder w/ top-K sparse attention for MI455X (gfx1250, wave32).
// GEMM-shaped math on v_wmma_f32_16x16x32_f16; GEMM B-panels staged to LDS
// with the Tensor Data Mover (tensor_load_to_lds + s_wait_tensorcnt), using
// TDM padding to de-conflict LDS banks.
// ---------------------------------------------------------------------------

typedef __attribute__((ext_vector_type(16))) _Float16 v16h;
typedef __attribute__((ext_vector_type(8)))  _Float16 h8;
typedef __attribute__((ext_vector_type(8)))  float    v8f;
typedef __attribute__((ext_vector_type(4)))  unsigned u32x4;
typedef __attribute__((ext_vector_type(8)))  int      i32x8;
typedef __attribute__((ext_vector_type(4)))  int      i32x4;

#define NLAYER 4
#define BB     2
#define SSEQ   1024
#define DDIM   512
#define HH     8
#define DK     64
#define KTOP   16
#define MROWS  (BB*SSEQ)     // 2048
#define LN_EPS 1e-6f
#define NEGV   -1000000000.0f

// LDS row stride for staged B panels: 512 halfs + 16 halfs (8 DWORD) TDM pad
// per row -> 1056 B stride (multiple of 16 B, breaks 256 B bank periodicity).
#define BROW   528

union FragH { v16h v; h8 h[2]; };

// A-matrix fragment (16xK row-major source, lda in halfs).
// ISA 16-bit A layout: lane row = lane%16; VGPR0..3 hold K=khalf..khalf+7,
// VGPR4..7 hold K=16+khalf..  with khalf = 8*(lane/16).
__device__ __forceinline__ v16h load_a_frag(const _Float16* base, int ld,
                                            int lr, int lh, int kk) {
  const _Float16* p = base + (size_t)lr * ld + kk + lh * 8;
  FragH f;
  f.h[0] = *(const h8*)p;        // K = kk+khalf .. +7
  f.h[1] = *(const h8*)(p + 16); // K = kk+16+khalf .. +7
  return f.v;
}

// B-matrix fragment from an N-major source Bsrc[n][k] (ld in halfs).
// B layout matching A: lane col = lane%16; VGPR0..3 hold K=khalf..khalf+7,
// VGPR4..7 hold K=16+khalf.. with khalf = 8*(lane/16).
__device__ __forceinline__ v16h load_b_frag(const _Float16* base, int ld,
                                            int lr, int lh, int kk) {
  const _Float16* p = base + (size_t)lr * ld + kk + lh * 8;
  FragH f;
  f.h[0] = *(const h8*)p;
  f.h[1] = *(const h8*)(p + 16);
  return f.v;
}

__device__ __forceinline__ v8f wmma32(v16h a, v16h b, v8f c) {
  return __builtin_amdgcn_wmma_f32_16x16x32_f16(false, a, false, b,
                                                (short)0, c, false, false);
}

// ---------------------------------------------------------------------------
// TDM: DMA a contiguous [rows x cols] f16 panel (row stride == cols in
// memory) into LDS, inserting 8 DWORDs of pad after every 256 DWORDs
// (= one 512-half row) so the LDS row stride becomes BROW halfs.
// Must be issued by a single wave; waits TENSORcnt==0 before returning.
// ---------------------------------------------------------------------------
#if __has_builtin(__builtin_amdgcn_tensor_load_to_lds)
#define HAVE_TDM 1
__device__ __forceinline__ void tdm_load_panel(_Float16* lds_dst,
                                               const _Float16* src,
                                               int rows, int cols) {
  const unsigned lds_off = (unsigned)(uintptr_t)lds_dst; // low 32b = LDS byte off
  const unsigned long long ga = (unsigned long long)(uintptr_t)src;

  u32x4 g0;
  g0[0] = 1u;                                        // count=1, user mode
  g0[1] = lds_off;                                   // lds_addr [63:32]
  g0[2] = (unsigned)(ga & 0xFFFFFFFFu);              // global_addr lo
  g0[3] = (unsigned)((ga >> 32) & 0x01FFFFFFu)       // global_addr [56:32]
        | (2u << 30);                                // type=2 ("image")

  i32x8 g1;
  g1[0] = (1 << 16)                                  // data_size = 2 bytes
        | (1 << 20)                                  // pad_enable
        | (7 << 22)                                  // pad_interval: 256 DWORDs
        | (7 << 25);                                 // pad_amount: 8 DWORDs
  g1[1] = (cols & 0xFFFF) << 16;                     // tensor_dim0 [15:0]
  g1[2] = ((cols >> 16) & 0xFFFF)                    // tensor_dim0 [31:16]
        | ((rows & 0xFFFF) << 16);                   // tensor_dim1 [15:0]
  g1[3] = ((rows >> 16) & 0xFFFF)                    // tensor_dim1 [31:16]
        | ((cols & 0xFFFF) << 16);                   // tile_dim0
  g1[4] = (rows & 0xFFFF);                           // tile_dim1 (tile_dim2=0)
  g1[5] = cols;                                      // tensor_dim0_stride lo
  g1[6] = 0;
  g1[7] = 0;

  i32x4 z4 = {0, 0, 0, 0};                           // groups 2/3 unused (2D)
#if __clang_major__ >= 23
  i32x8 z8 = {0, 0, 0, 0, 0, 0, 0, 0};
  __builtin_amdgcn_tensor_load_to_lds(g0, g1, z4, z4, z8, 0);
#else
  __builtin_amdgcn_tensor_load_to_lds(g0, g1, z4, z4, 0);
#endif
#if __has_builtin(__builtin_amdgcn_s_wait_tensorcnt)
  __builtin_amdgcn_s_wait_tensorcnt(0);
#else
  asm volatile("s_wait_tensorcnt 0x0" ::: "memory");
#endif
}
#else
#define HAVE_TDM 0
#endif

// ---------------------------------------------------------------------------
// One-time conversions
// ---------------------------------------------------------------------------

// Wt[(l*4+mat)*D*D + n*D + k] = W_mat[l][k][n]  (f32 -> f16, transposed)
__global__ __launch_bounds__(256)
void convert_w_kernel(const float* __restrict__ Wq, const float* __restrict__ Wk,
                      const float* __restrict__ Wv, const float* __restrict__ Wo,
                      _Float16* __restrict__ Wt) {
  size_t idx = (size_t)blockIdx.x * blockDim.x + threadIdx.x;
  int k   = (int)(idx % DDIM);
  int n   = (int)((idx / DDIM) % DDIM);
  int mat = (int)((idx / ((size_t)DDIM * DDIM)) % 4);
  int l   = (int)(idx / ((size_t)4 * DDIM * DDIM));
  const float* src = (mat == 0) ? Wq : (mat == 1) ? Wk : (mat == 2) ? Wv : Wo;
  float v = src[((size_t)l * DDIM + k) * DDIM + n];
  Wt[idx] = (_Float16)v;
}

__global__ __launch_bounds__(256)
void init_h_kernel(const float* __restrict__ x, float* __restrict__ hf32,
                   _Float16* __restrict__ hf16) {
  size_t i = (size_t)blockIdx.x * blockDim.x + threadIdx.x;
  float v = x[i];
  hf32[i] = v;
  hf16[i] = (_Float16)v;
}

// ---------------------------------------------------------------------------
// WMMA GEMM: C[M x 512] = A[M x 512] @ W[512 x 512] + bias (+ residual)
// A: f16 row-major (global, pipelined b128 loads). Bt: f16 N-major; the
// block's 128x512 B panel is TDM-staged into LDS with padded stride BROW.
// Each wave owns a 32x32 tile; block = 8 waves = 64(M) x 128(N).
// mode 0: f16 -> [B][H][S][DK]   (Q / K head layout)
// mode 2: f16 -> [B][H][DK][S]   (V transposed for PV B-fragments)
// mode 3: f32 y = c + bias + resid -> [M][D]   (O-projection)
// ---------------------------------------------------------------------------
__global__ __launch_bounds__(256)
void gemm_wmma_kernel(const _Float16* __restrict__ A, const _Float16* __restrict__ Bt,
                      const float* __restrict__ bias, const float* __restrict__ resid,
                      _Float16* __restrict__ dst16, float* __restrict__ dst32,
                      int mode) {
  extern __shared__ char smem_raw[];
  _Float16* s_b = (_Float16*)smem_raw;               // [128][BROW] halfs

  const int lane = threadIdx.x & 31;
  const int wave = threadIdx.x >> 5;
  const int lr = lane & 15, lh = lane >> 4;
  const int m0 = blockIdx.x * 64 + (wave >> 2) * 32;
  const int nblk = blockIdx.y * 128;
  const int n0l = (wave & 3) * 32;                   // n offset within LDS panel
  const int n0 = nblk + n0l;

  // ---- Stage B panel (contiguous 128KB in Bt) into LDS via the TDM ----
#if HAVE_TDM
  if (wave == 0) {
    tdm_load_panel(s_b, Bt + (size_t)nblk * DDIM, 128, DDIM);
  }
#else
  for (int i = threadIdx.x; i < 128 * (DDIM / 8); i += 256) {
    const int r = i / (DDIM / 8);
    const int c8 = i % (DDIM / 8);
    *(h8*)(s_b + (size_t)r * BROW + c8 * 8) =
        *(const h8*)(Bt + ((size_t)(nblk + r)) * DDIM + c8 * 8);
  }
#endif
  __syncthreads();

  v8f c00 = {0,0,0,0,0,0,0,0}, c01 = {0,0,0,0,0,0,0,0};
  v8f c10 = {0,0,0,0,0,0,0,0}, c11 = {0,0,0,0,0,0,0,0};

  const _Float16* Am0 = A + (size_t)m0 * DDIM;
  const _Float16* Am1 = A + (size_t)(m0 + 16) * DDIM;
  const _Float16* Bn0 = s_b + (size_t)n0l * BROW;
  const _Float16* Bn1 = s_b + (size_t)(n0l + 16) * BROW;

  for (int kk = 0; kk < DDIM; kk += 32) {
    __builtin_prefetch(Am0 + (size_t)lr * DDIM + kk + 32, 0, 0); // global_prefetch_b8
    v16h a0 = load_a_frag(Am0, DDIM, lr, lh, kk);
    v16h a1 = load_a_frag(Am1, DDIM, lr, lh, kk);
    v16h b0 = load_b_frag(Bn0, BROW, lr, lh, kk);    // ds_load from staged panel
    v16h b1 = load_b_frag(Bn1, BROW, lr, lh, kk);
    c00 = wmma32(a0, b0, c00);
    c01 = wmma32(a0, b1, c01);
    c10 = wmma32(a1, b0, c10);
    c11 = wmma32(a1, b1, c11);
  }

  for (int q = 0; q < 4; ++q) {
    v8f c = (q == 0) ? c00 : (q == 1) ? c01 : (q == 2) ? c10 : c11;
    const int mt = m0 + (q >> 1) * 16;
    const int nt = n0 + (q & 1) * 16;
    const int n = nt + lr;
    const float bval = bias[n];
    for (int r = 0; r < 8; ++r) {
      // C/D f32 layout: VGPR r -> M = r (lanes 0-15) or r+8 (lanes 16-31)
      const int m = mt + r + lh * 8;
      float v = c[r] + bval;
      const int b = m >> 10;           // m / SSEQ
      const int s = m & (SSEQ - 1);
      const int h = n >> 6;            // n / DK
      const int dk = n & (DK - 1);
      if (mode == 0) {
        dst16[(((size_t)(b * HH + h) * SSEQ + s) * DK) + dk] = (_Float16)v;
      } else if (mode == 2) {
        dst16[(((size_t)(b * HH + h) * DK + dk) * SSEQ) + s] = (_Float16)v;
      } else {
        float y = v + resid[(size_t)m * DDIM + n];
        dst32[(size_t)m * DDIM + n] = y;
      }
    }
  }
}

// ---------------------------------------------------------------------------
// Attention: block per (b, h, 16-query tile). 256 threads = 8 waves.
// LDS: scores f32 [16][1024] (64KB) + P f16 [16][1024] (32KB) = 96KB dynamic.
// Phase 1: scores = (Q @ K^T) * scale via WMMA          (8 waves x 8 key-tiles)
// Phase 2: per-row top-16 threshold (iterative wave-max extraction, handles
//          duplicates; keep = score >= kth & mask, matching jax semantics),
//          masked softmax -> f16 P in LDS                (wave per 2 rows)
// Phase 3: O = P @ V via WMMA from LDS                   (waves 0..3, d-tiles)
// ---------------------------------------------------------------------------
__global__ __launch_bounds__(256)
void attn_topk_kernel(const _Float16* __restrict__ Qh, const _Float16* __restrict__ Kh,
                      const _Float16* __restrict__ Vt, const unsigned char* __restrict__ mask,
                      _Float16* __restrict__ O16) {
  extern __shared__ char smem_raw[];
  float*    s_sc = (float*)smem_raw;                               // [16][1024]
  _Float16* s_p  = (_Float16*)(smem_raw + (size_t)16 * SSEQ * 4);  // [16][1024]

  const int qt = blockIdx.x, h = blockIdx.y, b = blockIdx.z;
  const int lane = threadIdx.x & 31, wave = threadIdx.x >> 5;
  const int lr = lane & 15, lh = lane >> 4;
  const float scale = 0.125f;  // 1/sqrt(DK)

  const _Float16* Qb = Qh + (size_t)(b * HH + h) * SSEQ * DK;
  const _Float16* Kb = Kh + (size_t)(b * HH + h) * SSEQ * DK;
  const _Float16* Vb = Vt + (size_t)(b * HH + h) * DK * SSEQ;

  // Q fragments for this 16-row query tile (DK=64 -> two K-steps of 32)
  const _Float16* Qrow = Qb + (size_t)qt * 16 * DK;
  v16h aq0 = load_a_frag(Qrow, DK, lr, lh, 0);
  v16h aq1 = load_a_frag(Qrow, DK, lr, lh, 32);

  // ---- Phase 1: scores ----
  for (int kt = wave; kt < SSEQ / 16; kt += 8) {
    const _Float16* Krow = Kb + (size_t)kt * 16 * DK;  // Bsrc[key][d], N-major
    v16h b0 = load_b_frag(Krow, DK, lr, lh, 0);
    v16h b1 = load_b_frag(Krow, DK, lr, lh, 32);
    v8f c = {0,0,0,0,0,0,0,0};
    c = wmma32(aq0, b0, c);
    c = wmma32(aq1, b1, c);
    for (int r = 0; r < 8; ++r) {
      const int m = r + lh * 8;
      const int key = kt * 16 + lr;
      s_sc[m * SSEQ + key] = c[r] * scale;
    }
  }
  __syncthreads();

  // ---- Phase 2: top-K threshold + masked softmax ----
  for (int row = wave; row < 16; row += 8) {
    float vals[32], work[32];
    for (int j = 0; j < 32; ++j) {
      vals[j] = s_sc[row * SSEQ + lane + 32 * j];
      work[j] = vals[j];
    }
    // 16th-largest raw score via iterative extraction (removes exactly one
    // instance per iteration -> correct with duplicates).
    float thr = NEGV;
    for (int it = 0; it < KTOP; ++it) {
      float lm = -3.0e38f; int lj = 0;
      for (int j = 0; j < 32; ++j) if (work[j] > lm) { lm = work[j]; lj = j; }
      float gm = lm;
      for (int off = 16; off > 0; off >>= 1)
        gm = fmaxf(gm, __shfl_xor(gm, off, 32));
      thr = gm;
      unsigned long long ball = __ballot(lm == gm);
      int first = (int)__ffsll(ball) - 1;
      if (lane == first) work[lj] = -3.0e38f;
    }
    // keep = (score >= thr) & mask ; masked-out -> NEG, then softmax
    const unsigned char* mrow = mask + (size_t)b * SSEQ;
    for (int j = 0; j < 32; ++j) {
      const int key = lane + 32 * j;
      const bool kp = (vals[j] >= thr) && (mrow[key] != 0);
      vals[j] = kp ? vals[j] : NEGV;
    }
    float mx = -3.0e38f;
    for (int j = 0; j < 32; ++j) mx = fmaxf(mx, vals[j]);
    for (int off = 16; off > 0; off >>= 1)
      mx = fmaxf(mx, __shfl_xor(mx, off, 32));
    float sum = 0.f;
    for (int j = 0; j < 32; ++j) sum += __expf(vals[j] - mx);
    for (int off = 16; off > 0; off >>= 1)
      sum += __shfl_xor(sum, off, 32);
    const float inv = 1.f / sum;
    for (int j = 0; j < 32; ++j) {
      const int key = lane + 32 * j;
      s_p[row * SSEQ + key] = (_Float16)(__expf(vals[j] - mx) * inv);
    }
  }
  __syncthreads();

  // ---- Phase 3: O = P @ V ----
  if (wave < 4) {
    const int n0 = wave * 16;  // d-tile within head
    v8f c = {0,0,0,0,0,0,0,0};
    const _Float16* Vrow = Vb + (size_t)n0 * SSEQ;  // Bsrc[d][key], N-major
    for (int kk = 0; kk < SSEQ; kk += 32) {
      v16h a  = load_a_frag(s_p, SSEQ, lr, lh, kk);   // LDS reads (ds_load)
      v16h bv = load_b_frag(Vrow, SSEQ, lr, lh, kk);
      c = wmma32(a, bv, c);
    }
    for (int r = 0; r < 8; ++r) {
      const int m = r + lh * 8;
      const int s = qt * 16 + m;
      const int d = h * DK + n0 + lr;
      O16[((size_t)(b * SSEQ + s)) * DDIM + d] = (_Float16)c[r];
    }
  }
}

// ---------------------------------------------------------------------------
// LayerNorm (reference quirk: beta scales, gamma shifts; std uses ddof=1).
// One wave per row of 512 floats; 8 rows per block.
// ---------------------------------------------------------------------------
__global__ __launch_bounds__(256)
void layernorm_kernel(const float* __restrict__ y, const float* __restrict__ gamma,
                      const float* __restrict__ beta, float* __restrict__ hf32,
                      _Float16* __restrict__ hf16, float* __restrict__ outF) {
  const int lane = threadIdx.x & 31, wave = threadIdx.x >> 5;
  const int row = blockIdx.x * 8 + wave;
  const float* yr = y + (size_t)row * DDIM;
  float v[16], s = 0.f, s2 = 0.f;
  for (int j = 0; j < 16; ++j) {
    v[j] = yr[lane + 32 * j];
    s += v[j];
    s2 += v[j] * v[j];
  }
  for (int off = 16; off > 0; off >>= 1) {
    s  += __shfl_xor(s, off, 32);
    s2 += __shfl_xor(s2, off, 32);
  }
  const float mean = s / (float)DDIM;
  const float var  = (s2 - (float)DDIM * mean * mean) / (float)(DDIM - 1);
  const float inv  = 1.f / (sqrtf(fmaxf(var, 0.f)) + LN_EPS);
  for (int j = 0; j < 16; ++j) {
    const int d = lane + 32 * j;
    const float o = beta[d] * (v[j] - mean) * inv + gamma[d];
    hf32[(size_t)row * DDIM + d] = o;
    hf16[(size_t)row * DDIM + d] = (_Float16)o;
    if (outF) outF[(size_t)row * DDIM + d] = o;
  }
}

// ---------------------------------------------------------------------------
// Host launcher
// ---------------------------------------------------------------------------
extern "C" void kernel_launch(void* const* d_in, const int* in_sizes, int n_in,
                              void* d_out, int out_size, void* d_ws, size_t ws_size,
                              hipStream_t stream) {
  (void)in_sizes; (void)n_in; (void)out_size; (void)ws_size;

  const float*         x     = (const float*)d_in[0];
  const unsigned char* mask  = (const unsigned char*)d_in[1];
  const float*         Wq    = (const float*)d_in[2];
  const float*         Wk    = (const float*)d_in[3];
  const float*         Wv    = (const float*)d_in[4];
  const float*         Wo    = (const float*)d_in[5];
  const float*         bq    = (const float*)d_in[6];
  const float*         bk    = (const float*)d_in[7];
  const float*         bv    = (const float*)d_in[8];
  const float*         bo    = (const float*)d_in[9];
  const float*         gamma = (const float*)d_in[10];
  const float*         beta  = (const float*)d_in[11];

  // Scratch carve-up (all sizes 256B-aligned by construction)
  char* ws = (char*)d_ws;
  _Float16* Wt   = (_Float16*)ws; ws += (size_t)NLAYER * 4 * DDIM * DDIM * 2; // 8 MB
  float*    hf32 = (float*)ws;    ws += (size_t)MROWS * DDIM * 4;             // 4 MB
  _Float16* hf16 = (_Float16*)ws; ws += (size_t)MROWS * DDIM * 2;             // 2 MB
  _Float16* Qh   = (_Float16*)ws; ws += (size_t)MROWS * DDIM * 2;             // 2 MB
  _Float16* Kh   = (_Float16*)ws; ws += (size_t)MROWS * DDIM * 2;             // 2 MB
  _Float16* Vt   = (_Float16*)ws; ws += (size_t)MROWS * DDIM * 2;             // 2 MB
  _Float16* O16  = (_Float16*)ws; ws += (size_t)MROWS * DDIM * 2;             // 2 MB
  float*    yf32 = (float*)ws;    ws += (size_t)MROWS * DDIM * 4;             // 4 MB

  // One-time conversions
  {
    const size_t nW = (size_t)NLAYER * 4 * DDIM * DDIM;
    convert_w_kernel<<<dim3((unsigned)(nW / 256)), dim3(256), 0, stream>>>(Wq, Wk, Wv, Wo, Wt);
    const size_t nH = (size_t)MROWS * DDIM;
    init_h_kernel<<<dim3((unsigned)(nH / 256)), dim3(256), 0, stream>>>(x, hf32, hf16);
  }

  const dim3 gemm_grid(MROWS / 64, DDIM / 128);
  const dim3 blk(256);
  const size_t gemm_smem = (size_t)128 * BROW * 2;                        // 132 KB
  const dim3 attn_grid(SSEQ / 16, HH, BB);
  const size_t attn_smem = (size_t)16 * SSEQ * 4 + (size_t)16 * SSEQ * 2; // 96 KB
  const dim3 ln_grid(MROWS / 8);

  for (int l = 0; l < NLAYER; ++l) {
    const _Float16* WtQ = Wt + (size_t)(l * 4 + 0) * DDIM * DDIM;
    const _Float16* WtK = Wt + (size_t)(l * 4 + 1) * DDIM * DDIM;
    const _Float16* WtV = Wt + (size_t)(l * 4 + 2) * DDIM * DDIM;
    const _Float16* WtO = Wt + (size_t)(l * 4 + 3) * DDIM * DDIM;

    gemm_wmma_kernel<<<gemm_grid, blk, gemm_smem, stream>>>(hf16, WtQ, bq + (size_t)l * DDIM,
                                                            nullptr, Qh, nullptr, 0);
    gemm_wmma_kernel<<<gemm_grid, blk, gemm_smem, stream>>>(hf16, WtK, bk + (size_t)l * DDIM,
                                                            nullptr, Kh, nullptr, 0);
    gemm_wmma_kernel<<<gemm_grid, blk, gemm_smem, stream>>>(hf16, WtV, bv + (size_t)l * DDIM,
                                                            nullptr, Vt, nullptr, 2);

    attn_topk_kernel<<<attn_grid, blk, attn_smem, stream>>>(Qh, Kh, Vt, mask, O16);

    gemm_wmma_kernel<<<gemm_grid, blk, gemm_smem, stream>>>(O16, WtO, bo + (size_t)l * DDIM,
                                                            hf32, nullptr, yf32, 3);

    float* extra = (l == NLAYER - 1) ? (float*)d_out : nullptr;
    layernorm_kernel<<<ln_grid, blk, 0, stream>>>(yf32, gamma + (size_t)l * DDIM,
                                                  beta + (size_t)l * DDIM,
                                                  hf32, hf16, extra);
  }
}
